// CATB_axial_24464133718791
// MI455X (gfx1250) — compile-verified, matrix operand order checked by reading, not run on hardware
//
#include <hip/hip_runtime.h>
#include <hip/hip_bf16.h>
#include <math.h>

typedef _Float16 half_t;
typedef _Float16 h8  __attribute__((ext_vector_type(8)));
typedef _Float16 v16h __attribute__((ext_vector_type(16)));
typedef float    v8f  __attribute__((ext_vector_type(8)));

#define B_      2
#define H_      112
#define W_      112
#define C_      192
#define L_      (H_*W_)
#define M_ROWS  (B_*L_)      // 25088
#define N_QKV   (3*C_)       // 576
#define HD      24
#define CB      96           // C/2 per branch
#define NTOK    784          // tokens per window (112*7)
#define RPOS    2899         // (2*112-1)*(2*7-1), same both branches
#define HIDDEN  (4*C_)       // 768

// ---------------------------------------------------------------- helpers ---

__device__ __forceinline__ v16h cat8(h8 a, h8 b) {
  return __builtin_shufflevector(a, b, 0,1,2,3,4,5,6,7,8,9,10,11,12,13,14,15);
}

__device__ __forceinline__ v8f wmma16(v16h a, v16h b, v8f c) {
  // D = A(16x32 f16) * B(32x16 f16) + C(16x16 f32)
  return __builtin_amdgcn_wmma_f32_16x16x32_f16(false, a, false, b, (short)0, c,
                                                false, false);
}

// A/B 16-bit operand load, ISA layout:
// lane<16 : k-halves {k0..k0+7, k0+16..k0+23}; lane>=16: {k0+8..15, k0+24..31}
// caller passes p0 = row_base + k0 + (lane>>4)*8
__device__ __forceinline__ v16h load_ab(const half_t* p0) {
  h8 lo = *(const h8*)p0;
  h8 hi = *(const h8*)(p0 + 16);
  return cat8(lo, hi);
}

// window token -> flat token index l in [0,L)
__device__ __forceinline__ int token_l(int br, int wid, int i) {
  if (br == 0) { int h = i / 7; int w = wid * 7 + (i - h * 7); return h * W_ + w; }
  return wid * NTOK + i;   // branch 1 windows are contiguous spans
}

// q row tile in A layout, head dim 24 padded to 32 with zeros
__device__ __forceinline__ v16h load_rowtile(const half_t* base, int row_l,
                                             int half_) {
  h8 z = {};
  const half_t* p = base + (size_t)row_l * N_QKV + half_ * 8;
  h8 lo = *(const h8*)p;                              // d in [0..15]
  h8 hi = (half_ == 0) ? *(const h8*)(p + 16) : z;    // d 16..23 valid, 24..31 pad
  return cat8(lo, hi);
}

// --------------------------------------------------------------- kernels ---

__global__ void cvt_f16_kernel(const float* __restrict__ in,
                               half_t* __restrict__ out, int n) {
  int i = blockIdx.x * blockDim.x + threadIdx.x;
  if (i < n) out[i] = (half_t)in[i];
}

// one wave per row LayerNorm, fp32 in -> f16 out
__global__ __launch_bounds__(256)
void ln_kernel(const float* __restrict__ in, const float* __restrict__ w,
               const float* __restrict__ b, half_t* __restrict__ out, int rows) {
  int lane = threadIdx.x & 31, wave = threadIdx.x >> 5;
  int row = blockIdx.x * 8 + wave;
  if (row >= rows) return;
  const float* p = in + (size_t)row * C_;
  float v[6], s = 0.f, s2 = 0.f;
#pragma unroll
  for (int i = 0; i < 6; ++i) { v[i] = p[lane + 32 * i]; s += v[i]; s2 += v[i] * v[i]; }
  for (int off = 16; off; off >>= 1) {
    s  += __shfl_xor(s,  off, 32);
    s2 += __shfl_xor(s2, off, 32);
  }
  float mu  = s * (1.f / C_);
  float inv = rsqrtf(fmaxf(s2 * (1.f / C_) - mu * mu, 0.f) + 1e-5f);
  half_t* o = out + (size_t)row * C_;
#pragma unroll
  for (int i = 0; i < 6; ++i) {
    int c = lane + 32 * i;
    o[c] = (half_t)((v[i] - mu) * inv * w[c] + b[c]);
  }
}

__device__ __forceinline__ void ln_relu6(float* v, const float* w, const float* b) {
  float mu = 0.f;
  for (int i = 0; i < 6; ++i) mu += v[i];
  mu *= (1.f / 6.f);
  float var = 0.f;
  for (int i = 0; i < 6; ++i) { float d = v[i] - mu; var += d * d; }
  var *= (1.f / 6.f);
  float inv = rsqrtf(var + 1e-5f);
  for (int i = 0; i < 6; ++i) v[i] = fmaxf((v[i] - mu) * inv * w[i] + b[i], 0.f);
}

// dynamic relative-position-bias MLP: 2 -> 6 -> 6 -> 6 -> 4, per table row
__global__ void pos_mlp_kernel(
    const float* __restrict__ proj_w, const float* __restrict__ proj_b,
    const float* __restrict__ ln1_w,  const float* __restrict__ ln1_b,
    const float* __restrict__ fc1_w,  const float* __restrict__ fc1_b,
    const float* __restrict__ ln2_w,  const float* __restrict__ ln2_b,
    const float* __restrict__ fc2_w,  const float* __restrict__ fc2_b,
    const float* __restrict__ ln3_w,  const float* __restrict__ ln3_b,
    const float* __restrict__ fc3_w,  const float* __restrict__ fc3_b,
    float* __restrict__ out /* [2][RPOS][4] */) {
  int t = blockIdx.x * blockDim.x + threadIdx.x;
  if (t >= 2 * RPOS) return;
  int br = t / RPOS, row = t - br * RPOS;
  int Hsp = br ? 7 : 112, Wsp = br ? 112 : 7;
  int bw = 2 * Wsp - 1;
  float ph = (float)(1 - Hsp + row / bw);
  float pw = (float)(1 - Wsp + row % bw);
  float v[6], u[6];
  for (int i = 0; i < 6; ++i)
    v[i] = proj_w[br * 12 + i * 2] * ph + proj_w[br * 12 + i * 2 + 1] * pw
         + proj_b[br * 6 + i];
  ln_relu6(v, ln1_w + br * 6, ln1_b + br * 6);
  for (int o = 0; o < 6; ++o) {
    float s = fc1_b[br * 6 + o];
    for (int i = 0; i < 6; ++i) s += fc1_w[br * 36 + o * 6 + i] * v[i];
    u[o] = s;
  }
  ln_relu6(u, ln2_w + br * 6, ln2_b + br * 6);
  for (int o = 0; o < 6; ++o) {
    float s = fc2_b[br * 6 + o];
    for (int i = 0; i < 6; ++i) s += fc2_w[br * 36 + o * 6 + i] * u[i];
    v[o] = s;
  }
  ln_relu6(v, ln3_w + br * 6, ln3_b + br * 6);
  for (int o = 0; o < 4; ++o) {
    float s = fc3_b[br * 4 + o];
    for (int i = 0; i < 6; ++i) s += fc3_w[br * 24 + o * 6 + i] * v[i];
    out[((size_t)br * RPOS + row) * 4 + o] = s;
  }
}

// generic f16 GEMM: out = act(A @ W^T + bias) [+ res], A: MxK f16, W: NxK f16
// flags: bit0 = f16 output, bit1 = exact GELU.  Requires K % 64 == 0.
__global__ __launch_bounds__(256)
void gemm_kernel(const half_t* __restrict__ A, const half_t* __restrict__ W,
                 const float* __restrict__ bias, const float* __restrict__ res,
                 void* __restrict__ out, int M, int N, int K, int flags) {
  int lane = threadIdx.x & 31, wave = threadIdx.x >> 5;
  int half_ = lane >> 4, l16 = lane & 15;
  int m0 = blockIdx.x * 128 + wave * 16;
  int n0 = blockIdx.y * 64;
  const half_t* arow = A + (size_t)(m0 + l16) * K + half_ * 8;
  const half_t* wrow = W + (size_t)(n0 + l16) * K + half_ * 8;
  v8f acc[4] = {};
  // ping-pong fragment sets (no register rotation)
  v16h a0 = load_ab(arow);
  v16h b0[4];
#pragma unroll
  for (int j = 0; j < 4; ++j) b0[j] = load_ab(wrow + (size_t)(16 * j) * K);
  for (int k0 = 0; k0 < K; k0 += 64) {
    v16h a1 = load_ab(arow + k0 + 32);
    v16h b1[4];
#pragma unroll
    for (int j = 0; j < 4; ++j)
      b1[j] = load_ab(wrow + (size_t)(16 * j) * K + k0 + 32);
    __builtin_prefetch(arow + k0 + 64, 0, 1);   // global_prefetch_b8
#pragma unroll
    for (int j = 0; j < 4; ++j) acc[j] = wmma16(a0, b0[j], acc[j]);
    int k2 = (k0 + 64 < K) ? k0 + 64 : 0;       // dummy in-bounds reload on last iter
    a0 = load_ab(arow + k2);
#pragma unroll
    for (int j = 0; j < 4; ++j) b0[j] = load_ab(wrow + (size_t)(16 * j) * K + k2);
#pragma unroll
    for (int j = 0; j < 4; ++j) acc[j] = wmma16(a1, b1[j], acc[j]);
  }
#pragma unroll
  for (int j = 0; j < 4; ++j) {
    int n = n0 + j * 16 + l16;
    float bv = bias ? bias[n] : 0.f;
#pragma unroll
    for (int r = 0; r < 8; ++r) {
      int m = m0 + r + 8 * half_;               // C/D layout: lane=N, vgpr=M
      float v = acc[j][r] + bv;
      if (flags & 2) v = 0.5f * v * (1.f + erff(v * 0.70710678118654752f));
      if (res) v += res[(size_t)m * N + n];
      if (flags & 1) ((half_t*)out)[(size_t)m * N + n] = (half_t)v;
      else           ((float*)out)[(size_t)m * N + n]  = v;
    }
  }
}

// flash attention: one workgroup = (batch, branch, window, head) x Q-round.
// 7 waves, one 16-row Q tile each; 25 key chunks of 32, double-buffered in LDS
// so chunk c+1 stages while chunk c computes (one barrier per chunk).
__global__ __launch_bounds__(224)
void attn_kernel(const half_t* __restrict__ qkvh, const float* __restrict__ postab,
                 half_t* __restrict__ attout) {
  __shared__ half_t pbuf[7][16][32]   __attribute__((aligned(16)));
  __shared__ half_t kbuf[2][32][32]   __attribute__((aligned(16)));
  __shared__ half_t vbuf[2][32][40]   __attribute__((aligned(16)));
  __shared__ float  poscol[RPOS];

  int bid   = blockIdx.x;
  int head  = bid & 3;
  int wid   = (bid >> 2) & 15;
  int br    = (bid >> 6) & 1;
  int batch = bid >> 7;

  for (int i = threadIdx.x; i < RPOS; i += 224)
    poscol[i] = postab[((size_t)br * RPOS + i) * 4 + head];

  int lane = threadIdx.x & 31, wave = threadIdx.x >> 5;
  int half_ = lane >> 4, l16 = lane & 15;
  int Hsp = br ? 7 : H_;
  int Wsp = br ? W_ : 7;
  int biasW = 2 * Wsp - 1;
  int ch = br * CB + head * HD;
  const half_t* qb = qkvh + (size_t)batch * L_ * N_QKV + ch;
  const half_t* kb = qb + C_;
  const half_t* vb = qb + 2 * C_;
  const float scale = 0.20412414523193154f;      // 24^-0.5

  int qt = blockIdx.y * 7 + wave;                // 0..48
  int i0 = qt * 16;
  v16h qa = load_rowtile(qb, token_l(br, wid, i0 + l16), half_);
  v8f o0 = {}, o1 = {};
  float mstat[8], lstat[8];
#pragma unroll
  for (int r = 0; r < 8; ++r) { mstat[r] = -1e30f; lstat[r] = 0.f; }

  int task = threadIdx.x;                        // staging tasks 0..63
  int trow = task & 31;
  const half_t* stage_base = (task & 32) ? vb : kb;

  // cooperative K/V chunk staging into LDS (coalesced, zero-padded)
  auto stage = [&](int cidx, int bufi) {
    if (task < 64) {
      int t = cidx * 32 + trow;
      h8 z = {};
      h8 r0 = z, r1 = z, r2 = z;
      if (t < NTOK) {
        const half_t* src = stage_base + (size_t)token_l(br, wid, t) * N_QKV;
        r0 = *(const h8*)src;
        r1 = *(const h8*)(src + 8);
        r2 = *(const h8*)(src + 16);
        if (t + 32 < NTOK)                       // prefetch the chunk after next
          __builtin_prefetch(stage_base + (size_t)token_l(br, wid, t + 32) * N_QKV,
                             0, 1);
      }
      half_t* dst = (task & 32) ? &vbuf[bufi][trow][0] : &kbuf[bufi][trow][0];
      *(h8*)dst        = r0;
      *(h8*)(dst + 8)  = r1;
      *(h8*)(dst + 16) = r2;                     // d 16..23
      *(h8*)(dst + 24) = z;                      // zero pad d 24..31
    }
  };

  stage(0, 0);
  __syncthreads();                               // poscol + first chunk ready

  for (int c = 0; c < 25; ++c) {
    int cur = c & 1;
    int j0 = c * 32;
    if (c + 1 < 25) stage(c + 1, cur ^ 1);       // overlap staging with compute

    // ---- S = Q K^T (two 16x16 tiles) ----
    const half_t* kp0 = &kbuf[cur][l16][half_ * 8];
    const half_t* kp1 = &kbuf[cur][16 + l16][half_ * 8];
    v16h kt0 = cat8(*(const h8*)kp0, *(const h8*)(kp0 + 16));
    v16h kt1 = cat8(*(const h8*)kp1, *(const h8*)(kp1 + 16));
    v8f zero = {};
    v8f s0 = wmma16(qa, kt0, zero);              // S[m][j0+0..15]
    v8f s1 = wmma16(qa, kt1, zero);              // S[m][j0+16..31]

    int jA = j0 + l16, jB = j0 + 16 + l16;
#pragma unroll
    for (int r = 0; r < 8; ++r) {
      int m = r + 8 * half_;
      int i = i0 + m;
      int hi = i / Wsp, wi = i - hi * Wsp;
      float v0, v1;
      if (jA < NTOK) {
        int hj = jA / Wsp, wj = jA - hj * Wsp;
        v0 = s0[r] * scale + poscol[(hi - hj + Hsp - 1) * biasW + (wi - wj + Wsp - 1)];
      } else v0 = -1e30f;
      if (jB < NTOK) {
        int hj = jB / Wsp, wj = jB - hj * Wsp;
        v1 = s1[r] * scale + poscol[(hi - hj + Hsp - 1) * biasW + (wi - wj + Wsp - 1)];
      } else v1 = -1e30f;

      float mx = fmaxf(v0, v1);                  // reduce within 16-lane half
      for (int off = 8; off; off >>= 1) mx = fmaxf(mx, __shfl_xor(mx, off, 32));
      float mnew  = fmaxf(mstat[r], mx);
      float alpha = __expf(mstat[r] - mnew);
      mstat[r] = mnew;
      float p0 = __expf(v0 - mnew);
      float p1 = __expf(v1 - mnew);
      float rs = p0 + p1;
      for (int off = 8; off; off >>= 1) rs += __shfl_xor(rs, off, 32);
      lstat[r] = lstat[r] * alpha + rs;
      o0[r] *= alpha;
      o1[r] *= alpha;
      pbuf[wave][m][l16]      = (half_t)p0;      // stage P for layout transpose
      pbuf[wave][m][16 + l16] = (half_t)p1;
    }

    // reload P in A layout (LDS ops from the same wave are in-order)
    const half_t* pp = &pbuf[wave][l16][half_ * 8];
    v16h pa = cat8(*(const h8*)pp, *(const h8*)(pp + 16));

    // V B-tiles from LDS: B[k=t][n=d]; pad columns are zero => branch-free
    v16h vt0, vt1;
    int d0 = l16, d1 = 16 + l16;
#pragma unroll
    for (int e = 0; e < 8; ++e) {
      int t = half_ * 8 + e;
      vt0[e]     = vbuf[cur][t][d0];
      vt0[8 + e] = vbuf[cur][t + 16][d0];
      vt1[e]     = vbuf[cur][t][d1];
      vt1[8 + e] = vbuf[cur][t + 16][d1];
    }
    o0 = wmma16(pa, vt0, o0);
    o1 = wmma16(pa, vt1, o1);
    __syncthreads();                             // next chunk staged & buffers free
  }

#pragma unroll
  for (int r = 0; r < 8; ++r) {
    int m = r + 8 * half_;
    int l = token_l(br, wid, i0 + m);
    float inv = 1.f / lstat[r];
    half_t* orow = attout + ((size_t)batch * L_ + l) * C_ + ch;
    orow[l16] = (half_t)(o0[r] * inv);
    if (l16 < 8) orow[16 + l16] = (half_t)(o1[r] * inv);
  }
}

// ----------------------------------------------------------------- launch ---

extern "C" void kernel_launch(void* const* d_in, const int* in_sizes, int n_in,
                              void* d_out, int out_size, void* d_ws, size_t ws_size,
                              hipStream_t stream) {
  const float* x          = (const float*)d_in[0];
  const float* norm1_w    = (const float*)d_in[1];
  const float* norm1_b    = (const float*)d_in[2];
  const float* qkv_w      = (const float*)d_in[3];
  const float* proj_w     = (const float*)d_in[4];
  const float* proj_b     = (const float*)d_in[5];
  const float* norm2_w    = (const float*)d_in[6];
  const float* norm2_b    = (const float*)d_in[7];
  const float* fc1_w      = (const float*)d_in[8];
  const float* fc1_b      = (const float*)d_in[9];
  const float* fc2_w      = (const float*)d_in[10];
  const float* fc2_b      = (const float*)d_in[11];
  const float* pos_proj_w = (const float*)d_in[12];
  const float* pos_proj_b = (const float*)d_in[13];
  const float* pos_ln1_w  = (const float*)d_in[14];
  const float* pos_ln1_b  = (const float*)d_in[15];
  const float* pos_fc1_w  = (const float*)d_in[16];
  const float* pos_fc1_b  = (const float*)d_in[17];
  const float* pos_ln2_w  = (const float*)d_in[18];
  const float* pos_ln2_b  = (const float*)d_in[19];
  const float* pos_fc2_w  = (const float*)d_in[20];
  const float* pos_fc2_b  = (const float*)d_in[21];
  const float* pos_ln3_w  = (const float*)d_in[22];
  const float* pos_ln3_b  = (const float*)d_in[23];
  const float* pos_fc3_w  = (const float*)d_in[24];
  const float* pos_fc3_b  = (const float*)d_in[25];
  (void)in_sizes; (void)n_in; (void)out_size; (void)ws_size;

  char* wsp = (char*)d_ws;
  size_t off = 0;
  auto alloc = [&](size_t bytes) -> void* {
    void* p = wsp + off;
    off = (off + bytes + 255) & ~(size_t)255;
    return p;
  };
  half_t* imgh  = (half_t*)alloc((size_t)M_ROWS * C_ * 2);
  half_t* qkvh  = (half_t*)alloc((size_t)M_ROWS * N_QKV * 2);
  half_t* atth  = (half_t*)alloc((size_t)M_ROWS * C_ * 2);
  float*  xa    = (float*) alloc((size_t)M_ROWS * C_ * 4);
  half_t* yh    = (half_t*)alloc((size_t)M_ROWS * C_ * 2);
  half_t* h1h   = (half_t*)alloc((size_t)M_ROWS * HIDDEN * 2);
  half_t* wqkv  = (half_t*)alloc((size_t)N_QKV * C_ * 2);
  half_t* wproj = (half_t*)alloc((size_t)C_ * C_ * 2);
  half_t* wfc1  = (half_t*)alloc((size_t)HIDDEN * C_ * 2);
  half_t* wfc2  = (half_t*)alloc((size_t)C_ * HIDDEN * 2);
  float*  ptab  = (float*) alloc((size_t)2 * RPOS * 4 * 4);

  // weight conversion to f16
  cvt_f16_kernel<<<(N_QKV * C_ + 255) / 256, 256, 0, stream>>>(qkv_w, wqkv, N_QKV * C_);
  cvt_f16_kernel<<<(C_ * C_ + 255) / 256, 256, 0, stream>>>(proj_w, wproj, C_ * C_);
  cvt_f16_kernel<<<(HIDDEN * C_ + 255) / 256, 256, 0, stream>>>(fc1_w, wfc1, HIDDEN * C_);
  cvt_f16_kernel<<<(C_ * HIDDEN + 255) / 256, 256, 0, stream>>>(fc2_w, wfc2, C_ * HIDDEN);

  // relative position bias table
  pos_mlp_kernel<<<(2 * RPOS + 255) / 256, 256, 0, stream>>>(
      pos_proj_w, pos_proj_b, pos_ln1_w, pos_ln1_b, pos_fc1_w, pos_fc1_b,
      pos_ln2_w, pos_ln2_b, pos_fc2_w, pos_fc2_b, pos_ln3_w, pos_ln3_b,
      pos_fc3_w, pos_fc3_b, ptab);

  // LN1 -> f16
  ln_kernel<<<M_ROWS / 8, 256, 0, stream>>>(x, norm1_w, norm1_b, imgh, M_ROWS);

  // QKV GEMM (f16 out)
  gemm_kernel<<<dim3(M_ROWS / 128, N_QKV / 64), 256, 0, stream>>>(
      imgh, wqkv, nullptr, nullptr, qkvh, M_ROWS, N_QKV, C_, /*flags=*/1);

  // flash attention: grid.x = batch/branch/window/head, grid.y = Q-tile round
  attn_kernel<<<dim3(256, 7), 224, 0, stream>>>(qkvh, ptab, atth);

  // proj GEMM + bias + residual(x) -> xa (f32)
  gemm_kernel<<<dim3(M_ROWS / 128, C_ / 64), 256, 0, stream>>>(
      atth, wproj, proj_b, x, xa, M_ROWS, C_, C_, /*flags=*/0);

  // LN2 -> f16
  ln_kernel<<<M_ROWS / 8, 256, 0, stream>>>(xa, norm2_w, norm2_b, yh, M_ROWS);

  // fc1 GEMM + bias + GELU (f16 out)
  gemm_kernel<<<dim3(M_ROWS / 128, HIDDEN / 64), 256, 0, stream>>>(
      yh, wfc1, fc1_b, nullptr, h1h, M_ROWS, HIDDEN, C_, /*flags=*/3);

  // fc2 GEMM + bias + residual(xa) -> d_out (f32)
  gemm_kernel<<<dim3(M_ROWS / 128, C_ / 64), 256, 0, stream>>>(
      h1h, wfc2, fc2_b, xa, (float*)d_out, M_ROWS, C_, HIDDEN, /*flags=*/0);
}
